// MoELayer_27462020891219
// MI455X (gfx1250) — matrix operand order repaired and checked by reading, not compiled, not run on hardware
//
#include <hip/hip_runtime.h>
#include <hip/hip_bf16.h>
#include <math.h>

typedef __bf16 bf16;
typedef __attribute__((ext_vector_type(16))) __bf16 v16bf;
typedef __attribute__((ext_vector_type(8)))  float  v8f;

#define T_TOK 2048
#define H_DIM 1024
#define I_DIM 2816
#define E_NUM 8
#define CAP   2048            // per-expert token capacity
#define SLOTS 4096            // T * K

// ---- workspace layout (bytes) ----
#define OFS_COUNTS 0
#define OFS_BASE   64
#define OFS_PSUM   128
#define OFS_FSUM   192
#define OFS_ZPAD   256        // 64B of zeros (async source for invalid rows)
#define OFS_ETOK   320
#define OFS_REC    (OFS_ETOK + E_NUM * CAP * 4)
#define OFS_WT     (OFS_REC + T_TOK * 2 * 4)
#define OFS_XB     (((OFS_WT + T_TOK * 2 * 4) + 255) & ~255)
#define OFS_ACT    (OFS_XB + T_TOK * H_DIM * 2)
#define OFS_SOUT   (OFS_ACT + (size_t)SLOTS * I_DIM * 2)

// ---- CDNA5 async global->LDS copy (ASYNCcnt path) ----
__device__ __forceinline__ void async_copy16(void* lds_dst, const void* gsrc) {
    // LDS byte offset = low 32 bits of the flat shared-memory address (ISA 10.2)
    unsigned lo = (unsigned)(size_t)lds_dst;
    asm volatile("global_load_async_to_lds_b128 %0, %1, off"
                 :: "v"(lo), "v"(gsrc) : "memory");
}
__device__ __forceinline__ void async_wait0() {
    asm volatile("s_wait_asynccnt 0x0" ::: "memory");
}

// ---------------- kernel 0: clear small state ----------------
__global__ void k_zero(int* counts, float* psum, float* fsum, int* zpad) {
    int i = threadIdx.x;
    if (i < E_NUM) { counts[i] = 0; psum[i] = 0.0f; fsum[i] = 0.0f; }
    if (i < 16) zpad[i] = 0;
}

// ------- kernel 1: LayerNorm + router + top-2 compaction + aux partials -------
__global__ void __launch_bounds__(256) k_ln_router(
    const float* __restrict__ x, const float* __restrict__ rW,
    const float* __restrict__ gamma, const float* __restrict__ beta,
    bf16* __restrict__ xb, int* counts, int* etok, int* recs, float* wts,
    float* psum, float* fsum)
{
    __shared__ float red[256];
    __shared__ float slog[E_NUM];
    const int t = blockIdx.x, tid = threadIdx.x;
    const float* xr = x + (size_t)t * H_DIM;

    float v[4]; float s = 0.0f;
    #pragma unroll
    for (int i = 0; i < 4; ++i) { v[i] = xr[tid + i * 256]; s += v[i]; }
    red[tid] = s; __syncthreads();
    for (int o = 128; o > 0; o >>= 1) { if (tid < o) red[tid] += red[tid + o]; __syncthreads(); }
    float mu = red[0] * (1.0f / H_DIM);
    __syncthreads();

    float s2 = 0.0f;
    #pragma unroll
    for (int i = 0; i < 4; ++i) { float d = v[i] - mu; s2 += d * d; }
    red[tid] = s2; __syncthreads();
    for (int o = 128; o > 0; o >>= 1) { if (tid < o) red[tid] += red[tid + o]; __syncthreads(); }
    float rstd = rsqrtf(red[0] * (1.0f / H_DIM) + 1e-5f);
    __syncthreads();

    float plog[E_NUM] = {};
    #pragma unroll
    for (int i = 0; i < 4; ++i) {
        int h = tid + i * 256;
        float nv = (v[i] - mu) * rstd * gamma[h] + beta[h];
        xb[(size_t)t * H_DIM + h] = (bf16)nv;
        #pragma unroll
        for (int e = 0; e < E_NUM; ++e) plog[e] += nv * rW[h * E_NUM + e];
    }
    for (int e = 0; e < E_NUM; ++e) {
        red[tid] = plog[e]; __syncthreads();
        for (int o = 128; o > 0; o >>= 1) { if (tid < o) red[tid] += red[tid + o]; __syncthreads(); }
        if (tid == 0) slog[e] = red[0];
        __syncthreads();
    }

    if (tid == 0) {
        float l[E_NUM];
        #pragma unroll
        for (int e = 0; e < E_NUM; ++e) l[e] = slog[e];
        int i0 = 0;
        for (int e = 1; e < E_NUM; ++e) if (l[e] > l[i0]) i0 = e;
        int i1 = (i0 == 0) ? 1 : 0;
        for (int e = 0; e < E_NUM; ++e) if (e != i0 && l[e] > l[i1]) i1 = e;
        // full softmax for aux loss
        float m = l[i0], se = 0.0f, pe[E_NUM];
        #pragma unroll
        for (int e = 0; e < E_NUM; ++e) { pe[e] = __expf(l[e] - m); se += pe[e]; }
        float inv = 1.0f / se;
        #pragma unroll
        for (int e = 0; e < E_NUM; ++e) atomicAdd(&psum[e], pe[e] * inv);
        atomicAdd(&fsum[i0], 1.0f);
        atomicAdd(&fsum[i1], 1.0f);
        // renormalized top-2 weights
        float e1 = __expf(l[i1] - l[i0]);
        float w0 = 1.0f / (1.0f + e1), w1 = e1 / (1.0f + e1);
        int idx0 = atomicAdd(&counts[i0], 1);
        etok[i0 * CAP + idx0] = t;
        recs[t * 2 + 0] = (i0 << 16) | idx0; wts[t * 2 + 0] = w0;
        int idx1 = atomicAdd(&counts[i1], 1);
        etok[i1 * CAP + idx1] = t;
        recs[t * 2 + 1] = (i1 << 16) | idx1; wts[t * 2 + 1] = w1;
    }
}

// ---------------- kernel 2: scan bases + aux loss ----------------
__global__ void k_scan_aux(const int* counts, int* bases,
                           const float* psum, const float* fsum, float* aux_out)
{
    if (threadIdx.x == 0) {
        int acc = 0;
        for (int e = 0; e < E_NUM; ++e) { bases[e] = acc; acc += counts[e]; }
        float s = 0.0f;
        const float invT = 1.0f / (float)T_TOK;
        for (int e = 0; e < E_NUM; ++e) s += (fsum[e] * invT) * (psum[e] * invT);
        aux_out[0] = 0.01f * (float)E_NUM * s;
    }
}

// ---------------- kernel 3: grouped gate/up GEMM + SwiGLU ----------------
__global__ void __launch_bounds__(128) k_gateup(
    const bf16* __restrict__ xb, const float* __restrict__ wg, const float* __restrict__ wu,
    const int* counts, const int* bases, const int* etok,
    const bf16* __restrict__ zpad, bf16* __restrict__ act)
{
    const int e = blockIdx.z, mt = blockIdx.y, nt = blockIdx.x;
    const int cnt = counts[e];
    if (mt * 64 >= cnt) return;
    const int base = bases[e];
    const int n0 = nt * 64;

    __shared__ bf16 sA[64][40];   // [m][k]  (row stride 80B = 5*16, 16B aligned)
    __shared__ bf16 sBg[64][40];  // transposed: [n][k]
    __shared__ bf16 sBu[64][40];
    __shared__ int stok[64];

    const int tid = threadIdx.x;
    if (tid < 64) {
        int r = mt * 64 + tid;
        stok[tid] = (r < cnt) ? etok[e * CAP + r] : -1;
    }
    __syncthreads();

    const int wid = tid >> 5, lane = tid & 31;
    const int lm = lane & 15, lh = lane >> 4;

    // per-thread staging roles (fixed across K loop)
    const int arow = tid >> 1, ac0 = (tid & 1) * 16;
    const int atok = stok[arow];
    const int bkk = tid >> 2, bnq = (tid & 3) * 16;

    v8f accG[4] = {}; v8f accU[4] = {};
    const float* wgE = wg + (size_t)e * H_DIM * I_DIM;
    const float* wuE = wu + (size_t)e * H_DIM * I_DIM;

    for (int k0 = 0; k0 < H_DIM; k0 += 32) {
        // stage A tile 64x32 bf16: async global->LDS (16B per lane)
        {
            const void* src = (atok >= 0)
                ? (const void*)(xb + (size_t)atok * H_DIM + k0 + ac0)
                : (const void*)zpad;
            async_copy16(&sA[arow][ac0], src);
        }
        // stage B tiles 32x64, f32 -> bf16, transposed into [n][k]
        {
            const float* pg = wgE + (size_t)(k0 + bkk) * I_DIM + n0 + bnq;
            const float* pu = wuE + (size_t)(k0 + bkk) * I_DIM + n0 + bnq;
            #pragma unroll
            for (int j = 0; j < 16; ++j) {
                sBg[bnq + j][bkk] = (bf16)pg[j];
                sBu[bnq + j][bkk] = (bf16)pu[j];
            }
            if (k0 + 32 < H_DIM) {
                __builtin_prefetch(pg + (size_t)32 * I_DIM, 0, 1);
                __builtin_prefetch(pu + (size_t)32 * I_DIM, 0, 1);
            }
        }
        async_wait0();
        __syncthreads();

        // A fragment: 16x32 bf16 per ISA layout (lane half -> K+8)
        v16bf a;
        {
            int m = wid * 16 + lm;
            #pragma unroll
            for (int v = 0; v < 8; ++v) {
                int k = ((v < 4) ? 0 : 16) + lh * 8 + 2 * (v & 3);
                a[2 * v]     = sA[m][k];
                a[2 * v + 1] = sA[m][k + 1];
            }
        }
        #pragma unroll
        for (int ns = 0; ns < 4; ++ns) {
            int n = ns * 16 + lm;
            int kb = lh * 8;
            v16bf bg, bu;
            #pragma unroll
            for (int v = 0; v < 8; ++v) {
                int kk = ((v < 4) ? 0 : 16) + kb + 2 * (v & 3);
                bg[2 * v]     = sBg[n][kk];
                bg[2 * v + 1] = sBg[n][kk + 1];
                bu[2 * v]     = sBu[n][kk];
                bu[2 * v + 1] = sBu[n][kk + 1];
            }
            accG[ns] = __builtin_amdgcn_wmma_f32_16x16x32_bf16(false, a, false, bg, (short)0, accG[ns], false, false);
            accU[ns] = __builtin_amdgcn_wmma_f32_16x16x32_bf16(false, a, false, bu, (short)0, accU[ns], false, false);
        }
        __syncthreads();
    }

    // epilogue: SiLU(gate) * up -> bf16 activations
    #pragma unroll
    for (int ns = 0; ns < 4; ++ns) {
        #pragma unroll
        for (int r = 0; r < 8; ++r) {
            int lmrow = wid * 16 + r + lh * 8;
            int tr = mt * 64 + lmrow;
            if (tr < cnt) {
                float g = accG[ns][r], u = accU[ns][r];
                float sg = g / (1.0f + __expf(-g));
                act[(size_t)(base + tr) * I_DIM + n0 + ns * 16 + lm] = (bf16)(sg * u);
            }
        }
    }
}

// ---------------- kernel 4: grouped down-projection GEMM ----------------
__global__ void __launch_bounds__(128) k_down(
    const bf16* __restrict__ act, const float* __restrict__ wd,
    const int* counts, const int* bases,
    const bf16* __restrict__ zpad, float* __restrict__ sout)
{
    const int e = blockIdx.z, mt = blockIdx.y, nt = blockIdx.x;
    const int cnt = counts[e];
    if (mt * 64 >= cnt) return;
    const int base = bases[e];
    const int n0 = nt * 64;

    __shared__ bf16 sA[64][40];
    __shared__ bf16 sB[64][40];  // transposed [n][k]

    const int tid = threadIdx.x;
    const int wid = tid >> 5, lane = tid & 31;
    const int lm = lane & 15, lh = lane >> 4;

    const int arow = tid >> 1, ac0 = (tid & 1) * 16;
    const bool avalid = (mt * 64 + arow) < cnt;
    const bf16* asrc = act + (size_t)(base + mt * 64 + arow) * I_DIM + ac0;
    const int bkk = tid >> 2, bnq = (tid & 3) * 16;

    v8f accD[4] = {};
    const float* wdE = wd + (size_t)e * I_DIM * H_DIM;

    for (int k0 = 0; k0 < I_DIM; k0 += 32) {
        // stage A tile: async global->LDS
        {
            const void* src = avalid ? (const void*)(asrc + k0) : (const void*)zpad;
            async_copy16(&sA[arow][ac0], src);
        }
        // stage B tile, f32 -> bf16, transposed
        {
            const float* pd = wdE + (size_t)(k0 + bkk) * H_DIM + n0 + bnq;
            #pragma unroll
            for (int j = 0; j < 16; ++j) sB[bnq + j][bkk] = (bf16)pd[j];
            if (k0 + 32 < I_DIM) __builtin_prefetch(pd + (size_t)32 * H_DIM, 0, 1);
        }
        async_wait0();
        __syncthreads();

        v16bf a;
        {
            int m = wid * 16 + lm;
            #pragma unroll
            for (int v = 0; v < 8; ++v) {
                int k = ((v < 4) ? 0 : 16) + lh * 8 + 2 * (v & 3);
                a[2 * v]     = sA[m][k];
                a[2 * v + 1] = sA[m][k + 1];
            }
        }
        #pragma unroll
        for (int ns = 0; ns < 4; ++ns) {
            int n = ns * 16 + lm;
            int kb = lh * 8;
            v16bf b;
            #pragma unroll
            for (int v = 0; v < 8; ++v) {
                int kk = ((v < 4) ? 0 : 16) + kb + 2 * (v & 3);
                b[2 * v]     = sB[n][kk];
                b[2 * v + 1] = sB[n][kk + 1];
            }
            accD[ns] = __builtin_amdgcn_wmma_f32_16x16x32_bf16(false, a, false, b, (short)0, accD[ns], false, false);
        }
        __syncthreads();
    }

    #pragma unroll
    for (int ns = 0; ns < 4; ++ns) {
        #pragma unroll
        for (int r = 0; r < 8; ++r) {
            int lmrow = wid * 16 + r + lh * 8;
            int tr = mt * 64 + lmrow;
            if (tr < cnt)
                sout[(size_t)(base + tr) * H_DIM + n0 + ns * 16 + lm] = accD[ns][r];
        }
    }
}

// ---------------- kernel 5: residual + weighted combine ----------------
__global__ void __launch_bounds__(256) k_combine(
    const float* __restrict__ x, const float* __restrict__ sout,
    const int* recs, const int* bases, const float* wts, float* __restrict__ out)
{
    const int t = blockIdx.x, tid = threadIdx.x;
    int r0 = recs[t * 2 + 0], r1 = recs[t * 2 + 1];
    int row0 = bases[r0 >> 16] + (r0 & 0xffff);
    int row1 = bases[r1 >> 16] + (r1 & 0xffff);
    float w0 = wts[t * 2 + 0], w1 = wts[t * 2 + 1];
    #pragma unroll
    for (int i = 0; i < 4; ++i) {
        int h = tid + i * 256;
        out[(size_t)t * H_DIM + h] = x[(size_t)t * H_DIM + h]
            + w0 * sout[(size_t)row0 * H_DIM + h]
            + w1 * sout[(size_t)row1 * H_DIM + h];
    }
}

extern "C" void kernel_launch(void* const* d_in, const int* in_sizes, int n_in,
                              void* d_out, int out_size, void* d_ws, size_t ws_size,
                              hipStream_t stream) {
    (void)in_sizes; (void)n_in; (void)out_size; (void)ws_size;
    const float* x     = (const float*)d_in[0];
    const float* rW    = (const float*)d_in[1];
    const float* wg    = (const float*)d_in[2];
    const float* wu    = (const float*)d_in[3];
    const float* wd    = (const float*)d_in[4];
    const float* gamma = (const float*)d_in[5];
    const float* beta  = (const float*)d_in[6];
    float* out = (float*)d_out;

    char* ws = (char*)d_ws;
    int*   counts = (int*)(ws + OFS_COUNTS);
    int*   bases  = (int*)(ws + OFS_BASE);
    float* psum   = (float*)(ws + OFS_PSUM);
    float* fsum   = (float*)(ws + OFS_FSUM);
    int*   zpad   = (int*)(ws + OFS_ZPAD);
    int*   etok   = (int*)(ws + OFS_ETOK);
    int*   recs   = (int*)(ws + OFS_REC);
    float* wts    = (float*)(ws + OFS_WT);
    bf16*  xb     = (bf16*)(ws + OFS_XB);
    bf16*  act    = (bf16*)(ws + OFS_ACT);
    float* sout   = (float*)(ws + OFS_SOUT);

    k_zero<<<dim3(1), dim3(64), 0, stream>>>(counts, psum, fsum, zpad);
    k_ln_router<<<dim3(T_TOK), dim3(256), 0, stream>>>(
        x, rW, gamma, beta, xb, counts, etok, recs, wts, psum, fsum);
    k_scan_aux<<<dim3(1), dim3(32), 0, stream>>>(
        counts, bases, psum, fsum, out + (size_t)T_TOK * H_DIM);
    k_gateup<<<dim3(I_DIM / 64, T_TOK / 64, E_NUM), dim3(128), 0, stream>>>(
        xb, wg, wu, counts, bases, etok, (const bf16*)zpad, act);
    k_down<<<dim3(H_DIM / 64, T_TOK / 64, E_NUM), dim3(128), 0, stream>>>(
        act, wd, counts, bases, (const bf16*)zpad, sout);
    k_combine<<<dim3(T_TOK), dim3(256), 0, stream>>>(
        x, sout, recs, bases, wts, out);
}
